// BiLSTMDCGenerator_88639535055032
// MI455X (gfx1250) — compile-verified, matrix operand order checked by reading, not run on hardware
//
#include <hip/hip_runtime.h>
#include <hip/hip_bf16.h>

typedef __attribute__((ext_vector_type(16))) _Float16 v16h;
typedef __attribute__((ext_vector_type(8)))  float    v8f;

#define DEVINL __device__ __forceinline__

// Fast nonlinearities: single v_exp_f32 + v_rcp_f32, correct saturation limits.
DEVINL float fast_sigm(float x){
  return __builtin_amdgcn_rcpf(1.0f + __expf(-x));
}
DEVINL float fast_tanh(float x){
  const float ax = __builtin_fabsf(x);
  const float t  = 1.0f - 2.0f * __builtin_amdgcn_rcpf(__expf(2.0f * ax) + 1.0f);
  return __builtin_copysignf(t, x);
}

DEVINL v8f wmma_f16(v16h a, v16h b, v8f c){
  // v_wmma_f32_16x16x32_f16: (neg_a, A, neg_b, B, c_mod, C, reuse_a, reuse_b)
  return __builtin_amdgcn_wmma_f32_16x16x32_f16(false, a, false, b, (short)0, c, false, false);
}

// ---- WMMA operand loaders (layouts per CDNA5 ISA 7.12.2) ----
// A tile 16x32 f16 from row-major [16][ld]: lane&15 = M; lanes>=16 shift K by 8.
DEVINL v16h load_a16(const _Float16* base, int ld, int lane, int k0){
  const int m  = lane & 15;
  const int kb = (lane >> 4) << 3;
  v16h a;
#pragma unroll
  for (int i = 0; i < 16; ++i){
    const int k = k0 + ((i >> 3) << 4) + kb + (i & 7);
    a[i] = base[m * ld + k];
  }
  return a;
}
// B tile 32x16, source stored [N][K] row-major (weight W[n][k]); B[k][n] = W[n][k]
DEVINL v16h load_b_nk(const _Float16* W, int ldk, int lane, int n0, int k0){
  const int n  = n0 + (lane & 15);
  const int kb = (lane >> 4) << 3;
  v16h b;
#pragma unroll
  for (int i = 0; i < 16; ++i){
    const int k = k0 + ((i >> 3) << 4) + kb + (i & 7);
    b[i] = W[n * ldk + k];
  }
  return b;
}
// B tile 32x16, source stored [K][N] row-major
DEVINL v16h load_b_kn(const _Float16* S, int ldn, int lane, int n0, int k0){
  const int n  = n0 + (lane & 15);
  const int kb = (lane >> 4) << 3;
  v16h b;
#pragma unroll
  for (int i = 0; i < 16; ++i){
    const int k = k0 + ((i >> 3) << 4) + kb + (i & 7);
    b[i] = S[k * ldn + n];
  }
  return b;
}
// C/D 16x16 f32: VGPR r -> M = r + 8*(lane>=16), N = lane&15
DEVINL v8f load_c(const float* base, int ld, int lane, int n0){
  const int n  = n0 + (lane & 15);
  const int mb = (lane >> 4) << 3;
  v8f c;
#pragma unroll
  for (int r = 0; r < 8; ++r) c[r] = base[(mb + r) * ld + n];
  return c;
}
DEVINL void store_c(float* base, int ld, int lane, int n0, v8f c){
  const int n  = n0 + (lane & 15);
  const int mb = (lane >> 4) << 3;
#pragma unroll
  for (int r = 0; r < 8; ++r) base[(mb + r) * ld + n] = c[r];
}

// ---- weight prep: f32 -> f16, zero-padded to (Rp, Cp) ----
__global__ void cvt_pad_f16(const float* __restrict__ src, _Float16* __restrict__ dst,
                            int R, int C, int Rp, int Cp){
  int i = blockIdx.x * blockDim.x + threadIdx.x;
  const int tot = Rp * Cp;
  for (; i < tot; i += gridDim.x * blockDim.x){
    const int r = i / Cp, c = i % Cp;
    const float v = (r < R && c < C) ? src[r * C + c] : 0.0f;
    dst[i] = (_Float16)v;
  }
}
// w2 torch layout (Cin=128, Cout=64, K=4) -> w2t[k][co][ci] f16 (WMMA A operand)
__global__ void prep_w2t(const float* __restrict__ w2, _Float16* __restrict__ dst){
  const int i = blockIdx.x * blockDim.x + threadIdx.x;
  if (i < 4 * 64 * 128){
    const int k  = i / (64 * 128);
    const int r  = i % (64 * 128);
    const int co = r / 128;
    const int ci = r % 128;
    dst[i] = (_Float16)w2[(ci * 64 + co) * 4 + k];
  }
}

// ================= Stage 1: small LSTM (in=5, hid=100), one direction =================
#define GLD1 404   // padded gate stride (bank-conflict offset for C store)
__global__ __launch_bounds__(256)
void lstm100_kernel(const float* __restrict__ x,
                    const float* __restrict__ h0, const float* __restrict__ c0,
                    const float* __restrict__ Wih, const float* __restrict__ bih,
                    const float* __restrict__ bhh,
                    const _Float16* __restrict__ Whh16,     // [400][128] f16, K-padded
                    _Float16* __restrict__ merged,          // (20,B,200) f16
                    int dir, int Bn)
{
  __shared__ float    sG[16 * GLD1];     // gates f32 (16 x 400)
  __shared__ _Float16 sH[16 * 128];      // h tile f16, K-padded
  __shared__ float    sWih[400 * 5];
  __shared__ float    sBias[400];

  const int tid  = threadIdx.x;
  const int lane = tid & 31;
  const int wave = tid >> 5;
  const int b0   = blockIdx.x * 16;

  for (int e = tid; e < 400 * 5; e += 256) sWih[e] = Wih[e];
  for (int e = tid; e < 400; e += 256) sBias[e] = bih[e] + bhh[e];
  for (int e = tid; e < 16 * 128; e += 256){
    const int m = e >> 7, k = e & 127;
    sH[e] = (k < 100) ? (_Float16)h0[(size_t)(b0 + m) * 100 + k] : (_Float16)0.0f;
  }
  float creg[7];
#pragma unroll
  for (int s = 0; s < 7; ++s){
    const int e = tid + 256 * s;
    creg[s] = (e < 1600) ? c0[(size_t)(b0 + e / 100) * 100 + (e % 100)] : 0.0f;
  }
  __syncthreads();

  for (int t = 0; t < 20; ++t){
    const int ts = dir ? (19 - t) : t;
    // gates = (bih+bhh) + x_t @ Wih^T   (K=5: VALU)
    for (int e = tid; e < 16 * 400; e += 256){
      const int m = e / 400, g = e % 400;
      const float* xr = x + ((size_t)ts * Bn + (b0 + m)) * 5;
      float s = sBias[g];
#pragma unroll
      for (int j = 0; j < 5; ++j) s += xr[j] * sWih[g * 5 + j];
      sG[m * GLD1 + g] = s;
    }
    __syncthreads();
    // gates += h @ Whh^T   (WMMA, K padded 100->128, N=400 -> 25 tiles over 8 waves)
    for (int nt = wave; nt < 25; nt += 8){
      const int n0 = nt * 16;
      __builtin_prefetch(Whh16 + (size_t)n0 * 128, 0, 1);
      v8f acc = load_c(sG, GLD1, lane, n0);
#pragma unroll
      for (int kc = 0; kc < 4; ++kc){
        v16h a = load_a16(sH, 128, lane, kc * 32);
        v16h b = load_b_nk(Whh16, 128, lane, n0, kc * 32);
        acc = wmma_f16(a, b, acc);
      }
      store_c(sG, GLD1, lane, n0, acc);
    }
    __syncthreads();
    // cell update (torch gate order i,f,g,o)
#pragma unroll
    for (int s = 0; s < 7; ++s){
      const int e = tid + 256 * s;
      if (e < 1600){
        const int m = e / 100, u = e % 100;
        const float gi = sG[m * GLD1 + u];
        const float gf = sG[m * GLD1 + 100 + u];
        const float gg = sG[m * GLD1 + 200 + u];
        const float go = sG[m * GLD1 + 300 + u];
        const float c2 = fast_sigm(gf) * creg[s] + fast_sigm(gi) * fast_tanh(gg);
        const float h2 = fast_sigm(go) * fast_tanh(c2);
        creg[s] = c2;
        sH[m * 128 + u] = (_Float16)h2;
        merged[((size_t)ts * Bn + (b0 + m)) * 200 + dir * 100 + u] = (_Float16)h2;
      }
    }
    __syncthreads();
  }
}

// ================= Stage 2: merge LSTM (200 -> 200) =================
#define GLD2 804
__global__ __launch_bounds__(256)
void lstm200_kernel(const _Float16* __restrict__ merged,
                    const float* __restrict__ h0, const float* __restrict__ c0,
                    const float* __restrict__ bih, const float* __restrict__ bhh,
                    const _Float16* __restrict__ Wih16,     // [800][224] f16
                    const _Float16* __restrict__ Whh16,     // [800][224] f16
                    _Float16* __restrict__ hm, int Bn)
{
  __shared__ float    sG[16 * GLD2];     // 51.5 KB gates
  __shared__ _Float16 sA[16 * 224];      // time-multiplexed h / x_t tile
  __shared__ float    sBias[800];

  const int tid  = threadIdx.x;
  const int lane = tid & 31;
  const int wave = tid >> 5;
  const int b0   = blockIdx.x * 16;

  for (int e = tid; e < 800; e += 256) sBias[e] = bih[e] + bhh[e];
  for (int e = tid; e < 16 * 224; e += 256){
    const int m = e / 224, k = e % 224;
    sA[e] = (k < 200) ? (_Float16)h0[(size_t)(b0 + m) * 200 + k] : (_Float16)0.0f;
  }
  float creg[13];
#pragma unroll
  for (int s = 0; s < 13; ++s){
    const int e = tid + 256 * s;
    creg[s] = (e < 3200) ? c0[(size_t)(b0 + e / 200) * 200 + (e % 200)] : 0.0f;
  }
  __syncthreads();

  for (int t = 0; t < 20; ++t){
    // phase A: gates = bias + h @ Whh^T
    for (int nt = wave; nt < 50; nt += 8){
      const int n0 = nt * 16;
      __builtin_prefetch(Whh16 + (size_t)n0 * 224, 0, 1);
      v8f acc;
      {
        const int n = n0 + (lane & 15);
#pragma unroll
        for (int r = 0; r < 8; ++r) acc[r] = sBias[n];
      }
#pragma unroll
      for (int kc = 0; kc < 7; ++kc){
        v16h a = load_a16(sA, 224, lane, kc * 32);
        v16h b = load_b_nk(Whh16, 224, lane, n0, kc * 32);
        acc = wmma_f16(a, b, acc);
      }
      store_c(sG, GLD2, lane, n0, acc);
    }
    __syncthreads();
    // overwrite sA with x_t (h no longer needed)
    for (int e = tid; e < 16 * 224; e += 256){
      const int m = e / 224, k = e % 224;
      sA[e] = (k < 200) ? merged[((size_t)t * Bn + (b0 + m)) * 200 + k] : (_Float16)0.0f;
    }
    __syncthreads();
    // phase B: gates += x_t @ Wih^T
    for (int nt = wave; nt < 50; nt += 8){
      const int n0 = nt * 16;
      __builtin_prefetch(Wih16 + (size_t)n0 * 224, 0, 1);
      v8f acc = load_c(sG, GLD2, lane, n0);
#pragma unroll
      for (int kc = 0; kc < 7; ++kc){
        v16h a = load_a16(sA, 224, lane, kc * 32);
        v16h b = load_b_nk(Wih16, 224, lane, n0, kc * 32);
        acc = wmma_f16(a, b, acc);
      }
      store_c(sG, GLD2, lane, n0, acc);
    }
    __syncthreads();
    // cell update; new h written into sA (pad cols stay zero)
#pragma unroll
    for (int s = 0; s < 13; ++s){
      const int e = tid + 256 * s;
      if (e < 3200){
        const int m = e / 200, u = e % 200;
        const float gi = sG[m * GLD2 + u];
        const float gf = sG[m * GLD2 + 200 + u];
        const float gg = sG[m * GLD2 + 400 + u];
        const float go = sG[m * GLD2 + 600 + u];
        const float c2 = fast_sigm(gf) * creg[s] + fast_sigm(gi) * fast_tanh(gg);
        const float h2 = fast_sigm(go) * fast_tanh(c2);
        creg[s] = c2;
        sA[m * 224 + u] = (_Float16)h2;
        if (t == 19) hm[(size_t)(b0 + m) * 200 + u] = (_Float16)h2;
      }
    }
    __syncthreads();
  }
}

// ========== Stage 3: dense + ConvT1/BN/ReLU + ConvT2(WMMA)/BN/ReLU + ConvT3 ==========
// One block = one batch row. ConvT stride2/pad1/k4: out l even=2j uses taps k=1(i=j),k=3(i=j-1);
// odd=2j+1 uses k=0(i=j+1),k=2(i=j). Guard columns make OOB taps zero.
__global__ __launch_bounds__(256)
void head_kernel(const _Float16* __restrict__ hm,
                 const float* __restrict__ Wd, const float* __restrict__ bd,
                 const float* __restrict__ w1, const float* __restrict__ b1,
                 const _Float16* __restrict__ w2t, const float* __restrict__ b2,
                 const float* __restrict__ w3, const float* __restrict__ b3,
                 const float* __restrict__ g1, const float* __restrict__ be1,
                 const float* __restrict__ m1, const float* __restrict__ v1,
                 const float* __restrict__ g2, const float* __restrict__ be2,
                 const float* __restrict__ m2, const float* __restrict__ v2,
                 float* __restrict__ out)
{
  __shared__ _Float16 sY1[128 * 72];   // y1[ci][pos], pos l stored at col l+4 (guards zero)
  __shared__ float    sY2[64 * 112];   // y2[co][pos], pos stored at col pos+2
  __shared__ float    sYb[32];         // dense out, index i stored at 1+i
  __shared__ float    sH[200];

  const int tid  = threadIdx.x;
  const int lane = tid & 31;
  const int wave = tid >> 5;
  const size_t row = blockIdx.x;

  for (int e = tid; e < 128 * 72; e += 256) sY1[e] = (_Float16)0.0f;
  for (int e = tid; e < 64 * 112; e += 256) sY2[e] = 0.0f;
  if (tid < 32) sYb[tid] = 0.0f;
  for (int e = tid; e < 200; e += 256) sH[e] = (float)hm[row * 200 + e];
  __syncthreads();

  // dense 200 -> 27
  if (tid < 27){
    float s = bd[tid];
    const float* wr = Wd + tid * 200;
    for (int k = 0; k < 200; ++k) s += sH[k] * wr[k];
    sYb[1 + tid] = s;
  }
  __syncthreads();

  // ConvT1 (1->128, 27->54) + BN1 + ReLU
  for (int e = tid; e < 128 * 54; e += 256){
    const int co = e / 54, l = e % 54;
    const int j = l >> 1;
    const float* wc = w1 + co * 4;
    float s = b1[co];
    if ((l & 1) == 0) s += sYb[1 + j] * wc[1] + sYb[j] * wc[3];
    else              s += sYb[2 + j] * wc[0] + sYb[1 + j] * wc[2];
    const float inv = g1[co] * rsqrtf(v1[co] + 1e-5f);
    s = (s - m1[co]) * inv + be1[co];
    sY1[co * 72 + (l + 4)] = (_Float16)fmaxf(s, 0.0f);
  }
  __syncthreads();

  // ConvT2 (128->64, 54->108) as WMMA GEMMs: M=co(64), K=ci(128), N=pos(54), 2 taps per parity.
  for (int it = 0; it < 4; ++it){
    const int job = wave + 8 * it;    // 32 tile jobs over 8 waves
    const int p   = job >> 4;         // output parity
    const int mt  = (job >> 2) & 3;   // co tile
    const int nt  = job & 3;          // j tile
    v8f acc;
    {
#pragma unroll
      for (int r = 0; r < 8; ++r) acc[r] = b2[mt * 16 + ((lane >> 4) << 3) + r];
    }
#pragma unroll
    for (int tap = 0; tap < 2; ++tap){
      const int kk  = (p == 0) ? (tap == 0 ? 1 : 3) : (tap == 0 ? 0 : 2);
      const int ofs = 4 + ((p == 0) ? (tap == 0 ? 0 : -1) : (tap == 0 ? 1 : 0));
      const _Float16* Abase = w2t + (size_t)kk * 64 * 128 + (size_t)mt * 16 * 128;
#pragma unroll
      for (int kc = 0; kc < 4; ++kc){
        v16h a = load_a16(Abase, 128, lane, kc * 32);
        v16h b = load_b_kn(sY1 + ofs, 72, lane, nt * 16, kc * 32);
        acc = wmma_f16(a, b, acc);
      }
    }
    const int j = nt * 16 + (lane & 15);
    if (j < 54){
#pragma unroll
      for (int r = 0; r < 8; ++r){
        const int co = mt * 16 + ((lane >> 4) << 3) + r;
        float s = acc[r];
        const float inv = g2[co] * rsqrtf(v2[co] + 1e-5f);
        s = (s - m2[co]) * inv + be2[co];
        sY2[co * 112 + (2 + 2 * j + p)] = fmaxf(s, 0.0f);
      }
    }
  }
  __syncthreads();

  // ConvT3 (64->1, 108->216)
  for (int l = tid; l < 216; l += 256){
    const int j = l >> 1;
    float s = b3[0];
    if ((l & 1) == 0){
      for (int ci = 0; ci < 64; ++ci)
        s += sY2[ci * 112 + 2 + j] * w3[ci * 4 + 1] + sY2[ci * 112 + 1 + j] * w3[ci * 4 + 3];
    } else {
      for (int ci = 0; ci < 64; ++ci)
        s += sY2[ci * 112 + 3 + j] * w3[ci * 4 + 0] + sY2[ci * 112 + 2 + j] * w3[ci * 4 + 2];
    }
    out[row * 216 + l] = s;
  }
}

extern "C" void kernel_launch(void* const* d_in, const int* in_sizes, int n_in,
                              void* d_out, int out_size, void* d_ws, size_t ws_size,
                              hipStream_t stream)
{
  (void)in_sizes; (void)n_in; (void)out_size; (void)ws_size;
  const int Bn = 16384;
  const float* x    = (const float*)d_in[0];
  const float* h0f  = (const float*)d_in[1];
  const float* c0f  = (const float*)d_in[2];
  const float* h0b  = (const float*)d_in[3];
  const float* c0b  = (const float*)d_in[4];
  const float* h0m  = (const float*)d_in[5];
  const float* c0m  = (const float*)d_in[6];
  const float* Wih_f= (const float*)d_in[7];
  const float* Whh_f= (const float*)d_in[8];
  const float* bih_f= (const float*)d_in[9];
  const float* bhh_f= (const float*)d_in[10];
  const float* Wih_b= (const float*)d_in[11];
  const float* Whh_b= (const float*)d_in[12];
  const float* bih_b= (const float*)d_in[13];
  const float* bhh_b= (const float*)d_in[14];
  const float* Wih_m= (const float*)d_in[15];
  const float* Whh_m= (const float*)d_in[16];
  const float* bih_m= (const float*)d_in[17];
  const float* bhh_m= (const float*)d_in[18];
  const float* Wd   = (const float*)d_in[19];
  const float* bd   = (const float*)d_in[20];
  const float* w1   = (const float*)d_in[21];
  const float* b1   = (const float*)d_in[22];
  const float* w2   = (const float*)d_in[23];
  const float* b2   = (const float*)d_in[24];
  const float* w3   = (const float*)d_in[25];
  const float* b3   = (const float*)d_in[26];
  const float* g1   = (const float*)d_in[27];
  const float* be1  = (const float*)d_in[28];
  const float* m1   = (const float*)d_in[29];
  const float* v1   = (const float*)d_in[30];
  const float* g2   = (const float*)d_in[31];
  const float* be2  = (const float*)d_in[32];
  const float* m2   = (const float*)d_in[33];
  const float* v2   = (const float*)d_in[34];

  char* ws = (char*)d_ws;
  size_t off = 0;
  auto take = [&](size_t bytes) -> char* {
    char* p = ws + off;
    off = (off + bytes + 255) & ~(size_t)255;
    return p;
  };
  _Float16* merged = (_Float16*)take((size_t)20 * Bn * 200 * 2);  // 131 MB
  _Float16* hm     = (_Float16*)take((size_t)Bn * 200 * 2);
  _Float16* WhhF16 = (_Float16*)take((size_t)400 * 128 * 2);
  _Float16* WhhB16 = (_Float16*)take((size_t)400 * 128 * 2);
  _Float16* WihM16 = (_Float16*)take((size_t)800 * 224 * 2);
  _Float16* WhhM16 = (_Float16*)take((size_t)800 * 224 * 2);
  _Float16* W2T    = (_Float16*)take((size_t)4 * 64 * 128 * 2);

  // weight prep (f32 -> f16, K-padded)
  cvt_pad_f16<<<dim3(256), dim3(256), 0, stream>>>(Whh_f, WhhF16, 400, 100, 400, 128);
  cvt_pad_f16<<<dim3(256), dim3(256), 0, stream>>>(Whh_b, WhhB16, 400, 100, 400, 128);
  cvt_pad_f16<<<dim3(512), dim3(256), 0, stream>>>(Wih_m, WihM16, 800, 200, 800, 224);
  cvt_pad_f16<<<dim3(512), dim3(256), 0, stream>>>(Whh_m, WhhM16, 800, 200, 800, 224);
  prep_w2t<<<dim3(128), dim3(256), 0, stream>>>(w2, W2T);

  // stage 1: bidirectional small LSTMs -> merged (20,B,200) f16
  lstm100_kernel<<<dim3(Bn / 16), dim3(256), 0, stream>>>(x, h0f, c0f, Wih_f, bih_f, bhh_f,
                                                          WhhF16, merged, 0, Bn);
  lstm100_kernel<<<dim3(Bn / 16), dim3(256), 0, stream>>>(x, h0b, c0b, Wih_b, bih_b, bhh_b,
                                                          WhhB16, merged, 1, Bn);
  // stage 2: merge LSTM -> hm (B,200) f16 (final carry only)
  lstm200_kernel<<<dim3(Bn / 16), dim3(256), 0, stream>>>(merged, h0m, c0m, bih_m, bhh_m,
                                                          WihM16, WhhM16, hm, Bn);
  // stage 3: fused dense + deconv head -> d_out (B,216) f32
  head_kernel<<<dim3(Bn), dim3(256), 0, stream>>>(hm, Wd, bd, w1, b1, W2T, b2, w3, b3,
                                                  g1, be1, m1, v1, g2, be2, m2, v2,
                                                  (float*)d_out);
}
